// BatchATSSAssigner_58420145160716
// MI455X (gfx1250) — compile-verified, matrix operand order checked by reading, not run on hardware
//
#include <hip/hip_runtime.h>
#include <hip/hip_bf16.h>

typedef __attribute__((ext_vector_type(16))) _Float16 v16h;
typedef __attribute__((ext_vector_type(8)))  float    v8f;

#define BS     32
#define NMAX   64
#define A_TOT  8400
#define NC     80
#define CHUNK  128   // anchors per LDS tile per wave

// -------- analytic anchors (generic/runtime version, used off the hot path) ----
__device__ __forceinline__ void anchor_info(int a, float& cx, float& cy,
                                            float& half, float& area) {
  int s, n, li;
  if (a < 6400)      { s = 8;  n = 80; li = a; }
  else if (a < 8000) { s = 16; n = 40; li = a - 6400; }
  else               { s = 32; n = 20; li = a - 8000; }
  int iy = li / n;
  int ix = li - iy * n;
  cx = (ix + 0.5f) * (float)s;
  cy = (iy + 0.5f) * (float)s;
  half = 2.5f * (float)s;
  float w = 5.0f * (float)s;
  area = w * w;
}

__device__ __forceinline__ float iou_gt_anchor(float gx1, float gy1, float gx2, float gy2,
                                               float garea, int a) {
  float cx, cy, hf, aarea;
  anchor_info(a, cx, cy, hf, aarea);
  float ax1 = cx - hf, ay1 = cy - hf, ax2 = cx + hf, ay2 = cy + hf;
  float iw = fmaxf(fminf(gx2, ax2) - fmaxf(gx1, ax1), 0.0f);
  float ih = fmaxf(fminf(gy2, ay2) - fmaxf(gy1, ay1), 0.0f);
  float inter = iw * ih;
  float uni = fmaxf(garea + aarea - inter, 1e-6f);
  return inter / uni;
}

// -------- K0: zero per-anchor counters / owners --------
__global__ void k0_init(int* fg_cnt, int* owner) {
  int t = blockIdx.x * blockDim.x + threadIdx.x;
  if (t < BS * A_TOT) { fg_cnt[t] = 0; owner[t] = 0; }
}

// -------- per-level pass: WMMA distance tiles + top-9 scan (compile-time level) ---
// S = stride, N = grid width, START/COUNT = anchor index range of this level.
// Division by N becomes mul-hi/shift since N is a template constant.
template<int S, int N, int START, int COUNT>
__device__ __forceinline__ void level_pass(const v16h& aF, float gsq,
                                           float* myLds, int m, int h,
                                           float (&d9)[9], int (&i9)[9]) {
  constexpr int NCH = (COUNT + CHUNK - 1) / CHUNK;
  for (int c = 0; c < NCH; ++c) {
    const int base = START + c * CHUNK;
    const int rem  = min(CHUNK, START + COUNT - base);

    __syncthreads();  // previous scan done before overwriting tiles

    // ---- WMMA phase: 8 tiles of 16 anchors; dot(m,n) = g_m . a_n ----
    for (int t = 0; t < CHUNK / 16; ++t) {
      v16h bF = {};  // B 32x16 f16: lanes 0-15 = N col, halves = K0..15
      const int colN = t * 16 + m;
      if (h == 0 && colN < rem) {
        int li = (base - START) + colN;
        int iy = li / N;          // constant divisor -> mul/shift
        int ix = li - iy * N;
        bF[0] = (_Float16)((ix + 0.5f) * (float)S);
        bF[1] = (_Float16)((iy + 0.5f) * (float)S);
      }
      v8f cF = {};
      cF = __builtin_amdgcn_wmma_f32_16x16x32_f16(
          false, aF, false, bF, (short)0, cF, false, false);
      // C layout: reg r, lane l -> (M = r + 8*(l>=16), N = l%16)
#pragma unroll
      for (int r = 0; r < 8; ++r)
        myLds[(r + h * 8) * CHUNK + t * 16 + m] = cF[r];
    }
    __syncthreads();

    // ---- scan phase: lane handles GT row m, cols [h*64, h*64+64) ----
    for (int j = 0; j < CHUNK / 2; ++j) {
      const int col = h * (CHUNK / 2) + j;
      if (col >= rem) break;
      int li = (base - START) + col;
      int iy = li / N;
      int ix = li - iy * N;
      float cx = (ix + 0.5f) * (float)S;
      float cy = (iy + 0.5f) * (float)S;
      float asq = cx * cx + cy * cy;
      const int aIdx = base + col;
      const float dot = myLds[m * CHUNK + col];
      const float d = fmaf(-2.0f, dot, gsq + asq);  // squared distance
      if (d < d9[8] || (d == d9[8] && aIdx < i9[8])) {
        int k = 8;
        while (k > 0 && (d < d9[k - 1] || (d == d9[k - 1] && aIdx < i9[k - 1]))) {
          d9[k] = d9[k - 1]; i9[k] = i9[k - 1]; --k;
        }
        d9[k] = d; i9[k] = aIdx;
      }
    }
  }
}

__device__ __forceinline__ void top9_init(float (&d9)[9], int (&i9)[9]) {
#pragma unroll
  for (int k = 0; k < 9; ++k) { d9[k] = 3.4e38f; i9[k] = 0x7fffffff; }
}

// merge the two lane halves (wave32 shuffle, width 32), write 9 indices
__device__ __forceinline__ void top9_merge(float (&d9)[9], int (&i9)[9],
                                           int lane, int h, int* out) {
  float od[9]; int oi[9];
#pragma unroll
  for (int k = 0; k < 9; ++k) {
    od[k] = __shfl(d9[k], lane + 16, 32);
    oi[k] = __shfl(i9[k], lane + 16, 32);
  }
  if (h == 0) {
    int p = 0, q = 0;
#pragma unroll
    for (int k = 0; k < 9; ++k) {
      bool t0 = (d9[p] < od[q]) || (d9[p] == od[q] && i9[p] <= oi[q]);
      out[k] = t0 ? i9[p] : oi[q];
      if (t0) ++p; else ++q;
    }
  }
}

// -------- K1: per-GT candidate selection via WMMA distance tiles --------
// grid = BS blocks, 128 threads (4 waves); wave w handles GTs [16w,16w+16)
__global__ __launch_bounds__(128) void k1_candidates(
    const float* __restrict__ gt_bboxes,  // (BS,NMAX,4)
    const float* __restrict__ mask_gt,    // (BS,NMAX,1)
    int* __restrict__ cand_idx,           // (BS*NMAX*27)
    int* __restrict__ cand_pos,           // (BS*NMAX*27)
    int* __restrict__ fg_cnt)             // (BS*A_TOT)
{
  __shared__ float ldsDot[4 * 16 * CHUNK];  // 32 KB: per-wave 16xCHUNK dot tiles

  const int b    = blockIdx.x;
  const int wave = threadIdx.x >> 5;
  const int lane = threadIdx.x & 31;
  const int m    = lane & 15;   // GT row within wave tile / N column for B frag
  const int h    = lane >> 4;   // half: which 64-col slice this lane scans
  const int gt   = wave * 16 + m;

  const float* gb = gt_bboxes + (size_t)(b * NMAX + gt) * 4;
  const float gx1 = gb[0], gy1 = gb[1], gx2 = gb[2], gy2 = gb[3];
  const float gcx = (gx1 + gx2) * 0.5f, gcy = (gy1 + gy2) * 0.5f;
  const float gsq = gcx * gcx + gcy * gcy;

  // A fragment: 16x32 f16, row M = lane (lanes 0-15), K0=gx K1=gy, rest 0
  v16h aF = {};
  if (h == 0) { aF[0] = (_Float16)gcx; aF[1] = (_Float16)gcy; }

  float* myLds = ldsDot + wave * 16 * CHUNK;

  int candIdx[27];
  float d9[9]; int i9[9];

  top9_init(d9, i9);
  level_pass<8, 80, 0, 6400>(aF, gsq, myLds, m, h, d9, i9);
  top9_merge(d9, i9, lane, h, candIdx + 0);

  top9_init(d9, i9);
  level_pass<16, 40, 6400, 1600>(aF, gsq, myLds, m, h, d9, i9);
  top9_merge(d9, i9, lane, h, candIdx + 9);

  top9_init(d9, i9);
  level_pass<32, 20, 8000, 400>(aF, gsq, myLds, m, h, d9, i9);
  top9_merge(d9, i9, lane, h, candIdx + 18);

  // ---- candidate thresholding (one lane per GT) ----
  if (h == 0) {
    const float garea = fmaxf(gx2 - gx1, 0.0f) * fmaxf(gy2 - gy1, 0.0f);
    const bool valid = mask_gt[b * NMAX + gt] > 0.0f;

    float o[27], sum = 0.0f;
    for (int j = 0; j < 27; ++j) {
      int a = candIdx[j];
      float v = 0.0f;
      if (valid && a >= 0 && a < A_TOT)
        v = iou_gt_anchor(gx1, gy1, gx2, gy2, garea, a);
      o[j] = v; sum += v;
    }
    const float mean = sum * (1.0f / 27.0f);
    float var = 0.0f;
    for (int j = 0; j < 27; ++j) { float e = o[j] - mean; var += e * e; }
    var *= (1.0f / 26.0f);  // ddof=1
    const float thr = mean + sqrtf(var);

    const int baseC = (b * NMAX + gt) * 27;
    for (int j = 0; j < 27; ++j) {
      int a = candIdx[j];
      int pos = 0;
      if (valid && a >= 0 && a < A_TOT && o[j] > thr) {
        float cx, cy, hf, ar;
        anchor_info(a, cx, cy, hf, ar);
        float dmin = fminf(fminf(cx - gx1, cy - gy1), fminf(gx2 - cx, gy2 - cy));
        if (dmin > 1e-9f) pos = 1;
      }
      cand_idx[baseC + j] = a;
      cand_pos[baseC + j] = pos;
      if (pos) atomicAdd(&fg_cnt[b * A_TOT + a], 1);
    }
  }
}

// -------- K2a: record owning GT for anchors claimed exactly once --------
__global__ void k2a_owner(const int* __restrict__ cand_idx,
                          const int* __restrict__ cand_pos,
                          const int* __restrict__ fg_cnt,
                          int* __restrict__ owner) {
  int t = blockIdx.x * blockDim.x + threadIdx.x;
  if (t >= BS * NMAX * 27) return;
  if (!cand_pos[t]) return;
  int bg = t / 27;
  int b = bg / NMAX, g = bg - b * NMAX;
  int a = cand_idx[t];
  int idx = b * A_TOT + a;
  if (fg_cnt[idx] == 1) owner[idx] = g;  // unique writer when cnt==1
}

// -------- K3a: conflict argmax + gathers + gt-vs-pd IoU --------
// 2D grid: x tiles anchors, y = batch. GT boxes of the batch are staged into
// LDS with the CDNA5 async-to-LDS path (ASYNCcnt, s_wait_asynccnt).
__global__ __launch_bounds__(256) void k3a_assign(
    const float* __restrict__ gt_bboxes,
    const int*   __restrict__ gt_labels,
    const float* __restrict__ pd_bboxes,
    const int*   __restrict__ fg_cnt,
    const int*   __restrict__ owner,
    float* __restrict__ lbl_out,
    float* __restrict__ bbox_out,
    float* __restrict__ fg_out,
    int*   __restrict__ ws_lbl,
    float* __restrict__ ws_iou) {
  __shared__ float4 shGt[NMAX];  // 1 KB: this batch's 64 GT boxes

  const int b = blockIdx.y;
  const int a = blockIdx.x * blockDim.x + threadIdx.x;

  // ---- async-stage GT boxes: waves 0-1 (full EXEC), 16B per lane ----
  if (threadIdx.x < NMAX) {
    unsigned int ldsOff = (unsigned int)(uintptr_t)(&shGt[threadIdx.x]);
    const float* gp = gt_bboxes + (size_t)(b * NMAX + threadIdx.x) * 4;
    asm volatile("global_load_async_to_lds_b128 %0, %1, off"
                 :: "v"(ldsOff), "v"(gp) : "memory");
  }
  asm volatile("s_wait_asynccnt 0x0" ::: "memory");
  __syncthreads();

  if (a >= A_TOT) return;
  const int idx = b * A_TOT + a;

  __builtin_prefetch(pd_bboxes + (size_t)idx * 4, 0, 0);  // global_prefetch_b8

  int cnt = fg_cnt[idx];
  int g = 0, fg = 0;
  if (cnt > 1) {
    fg = 1;
    float cx, cy, hf, aarea;
    anchor_info(a, cx, cy, hf, aarea);
    float ax1 = cx - hf, ay1 = cy - hf, ax2 = cx + hf, ay2 = cy + hf;
    float best = -1.0f;
    for (int gg = 0; gg < NMAX; ++gg) {
      float4 gbv = shGt[gg];
      float garea = fmaxf(gbv.z - gbv.x, 0.0f) * fmaxf(gbv.w - gbv.y, 0.0f);
      float iw = fmaxf(fminf(gbv.z, ax2) - fmaxf(gbv.x, ax1), 0.0f);
      float ih = fmaxf(fminf(gbv.w, ay2) - fmaxf(gbv.y, ay1), 0.0f);
      float inter = iw * ih;
      float iou = inter / fmaxf(garea + aarea - inter, 1e-6f);
      if (iou > best) { best = iou; g = gg; }  // first max wins (argmax)
    }
  } else if (cnt == 1) {
    fg = 1;
    g = owner[idx];
  }

  float4 gbv = shGt[g];
  int lbl = fg ? gt_labels[b * NMAX + g] : NC;
  float sc = 0.0f;
  if (fg) {
    float4 pb = ((const float4*)pd_bboxes)[idx];
    float a1 = fmaxf(gbv.z - gbv.x, 0.0f) * fmaxf(gbv.w - gbv.y, 0.0f);
    float a2 = fmaxf(pb.z - pb.x, 0.0f) * fmaxf(pb.w - pb.y, 0.0f);
    float iw = fmaxf(fminf(gbv.z, pb.z) - fmaxf(gbv.x, pb.x), 0.0f);
    float ih = fmaxf(fminf(gbv.w, pb.w) - fmaxf(gbv.y, pb.y), 0.0f);
    float inter = iw * ih;
    sc = inter / (a1 + a2 - inter + 1e-9f);
  }

  lbl_out[idx] = (float)lbl;
  ((float4*)bbox_out)[idx] = gbv;
  fg_out[idx] = fg ? 1.0f : 0.0f;
  ws_lbl[idx] = lbl;
  ws_iou[idx] = sc;
}

// -------- K3b: coalesced one-hot score expansion (86 MB, float4 stores) --------
__global__ void k3b_scores(const int* __restrict__ ws_lbl,
                           const float* __restrict__ ws_iou,
                           float* __restrict__ scores) {
  int t = blockIdx.x * blockDim.x + threadIdx.x;  // over BS*A_TOT*20
  if (t >= BS * A_TOT * (NC / 4)) return;
  int ba = t / (NC / 4);
  int sub = t - ba * (NC / 4);
  int c0 = sub * 4;
  int lbl = ws_lbl[ba];
  float sc = ws_iou[ba];
  float4 v;
  v.x = (lbl == c0    ) ? sc : 0.0f;
  v.y = (lbl == c0 + 1) ? sc : 0.0f;
  v.z = (lbl == c0 + 2) ? sc : 0.0f;
  v.w = (lbl == c0 + 3) ? sc : 0.0f;
  ((float4*)scores)[t] = v;
}

extern "C" void kernel_launch(void* const* d_in, const int* in_sizes, int n_in,
                              void* d_out, int out_size, void* d_ws, size_t ws_size,
                              hipStream_t stream) {
  (void)in_sizes; (void)n_in; (void)out_size; (void)ws_size;
  // inputs: 0=anc_bboxes(unused, analytic), 1=n_level_bboxes(unused),
  //         2=gt_labels, 3=gt_bboxes, 4=mask_gt, 5=pd_bboxes
  const int*   gt_labels = (const int*)d_in[2];
  const float* gt_bboxes = (const float*)d_in[3];
  const float* mask_gt   = (const float*)d_in[4];
  const float* pd_bboxes = (const float*)d_in[5];

  const int L0 = BS * A_TOT;  // 268800
  float* out    = (float*)d_out;
  float* lblO   = out;              // (BS,A)
  float* bboxO  = out + L0;         // (BS,A,4)
  float* scoreO = out + 5 * L0;     // (BS,A,80)
  float* fgO    = out + 85 * L0;    // (BS,A)

  int* cand_idx = (int*)d_ws;                      // BS*NMAX*27
  int* cand_pos = cand_idx + BS * NMAX * 27;
  int* fg_cnt   = cand_pos + BS * NMAX * 27;       // BS*A
  int* owner    = fg_cnt + L0;
  int* ws_lbl   = owner + L0;
  float* ws_iou = (float*)(ws_lbl + L0);

  k0_init<<<(L0 + 255) / 256, 256, 0, stream>>>(fg_cnt, owner);
  k1_candidates<<<BS, 128, 0, stream>>>(gt_bboxes, mask_gt, cand_idx, cand_pos, fg_cnt);
  k2a_owner<<<(BS * NMAX * 27 + 255) / 256, 256, 0, stream>>>(cand_idx, cand_pos, fg_cnt, owner);
  dim3 g3a((A_TOT + 255) / 256, BS);
  k3a_assign<<<g3a, 256, 0, stream>>>(gt_bboxes, gt_labels, pd_bboxes,
                                      fg_cnt, owner, lblO, bboxO, fgO,
                                      ws_lbl, ws_iou);
  k3b_scores<<<(L0 * (NC / 4) + 255) / 256, 256, 0, stream>>>(ws_lbl, ws_iou, scoreO);
}
